// LieFuncBeamSearchOptimized_30855045054565
// MI455X (gfx1250) — compile-verified, matrix operand order checked by reading, not run on hardware
//
#include <hip/hip_runtime.h>
#include <hip/hip_bf16.h>
#include <stdint.h>

// ---------------------------------------------------------------------------
// CDNA5 (gfx1250) implementation.
// All heavy GEMMs routed through v_wmma_f32_16x16x32_bf16 (wave32).
// GEMM A-tile staging uses async global->LDS copies (ASYNCcnt) when the
// toolchain exposes the gfx1250 builtins; otherwise falls back to VGPR staging.
// ---------------------------------------------------------------------------

typedef __attribute__((ext_vector_type(16))) __bf16 v16bf;
typedef __attribute__((ext_vector_type(8)))  float  v8f;
typedef __attribute__((ext_vector_type(4)))  int    v4i;

union FragBF {
  v16bf v;
  uint4 q[2];
};

#define WMMA_BF16(a, b, c) \
  __builtin_amdgcn_wmma_f32_16x16x32_bf16(false, (a), false, (b), (short)0, (c), false, false)

#if defined(__has_builtin)
#  if __has_builtin(__builtin_amdgcn_global_load_async_to_lds_b128) && \
      __has_builtin(__builtin_amdgcn_s_wait_asynccnt)
#    define USE_ASYNC_LDS 1
#  endif
#endif
#ifndef USE_ASYNC_LDS
#  define USE_ASYNC_LDS 0
#endif

#if USE_ASYNC_LDS
// signature (from compiler diagnostic): (v4i as1*, v4i as3*, imm offset, imm cpol)
typedef __attribute__((address_space(1))) v4i* gas_v4i_t;
typedef __attribute__((address_space(3))) v4i* las_v4i_t;
#endif

// ------------------------------ conversions --------------------------------

__global__ void k_cvt_bf16(const float* __restrict__ in, __bf16* __restrict__ out, int n) {
  int i = blockIdx.x * blockDim.x + threadIdx.x;
  if (i < n) out[i] = (__bf16)in[i];
}

// weight[0, :2048, :2048] with tril mask (keep row>=col), SEQ_MAX=4100 stride
__global__ void k_w0_tril(const float* __restrict__ w, __bf16* __restrict__ out) {
  int i = blockIdx.x * blockDim.x + threadIdx.x;   // 2048*2048
  int r = i >> 11, c = i & 2047;
  float v = (r >= c) ? w[(size_t)r * 4100 + c] : 0.0f;
  out[i] = (__bf16)v;
}

// gna_w1 [1024,512] f32 -> transposed bf16 [512][1024]
__global__ void k_w1t(const float* __restrict__ w, __bf16* __restrict__ out) {
  int i = blockIdx.x * blockDim.x + threadIdx.x;   // 1024*512
  int k = i >> 9, n = i & 511;
  out[(size_t)n * 1024 + k] = (__bf16)w[i];
}

// ------------------------------ generic GEMM -------------------------------
// C[M,N] = act( A[M,K] @ B[K,N] + bias )
// A element (m,k) address: (m/MB)*sB + (m%MB)*sM + k*sK   (handles ct transpose)
// Tile 128x128, 8 waves, each wave computes 32x64 (2x4 WMMA accumulators).

__global__ __launch_bounds__(256) void gemm_bf16_kernel(
    const __bf16* __restrict__ A, const __bf16* __restrict__ Bm,
    const float* __restrict__ bias, void* __restrict__ C,
    int M, int N, int K, int MB, long sB, long sM, long sK,
    int relu, int outbf16)
{
  __shared__ __bf16 lA[128][32];
  __shared__ __bf16 lBt[128][32];

  const int tid  = threadIdx.x;
  const int lane = tid & 31;
  const int wv   = tid >> 5;
  const int wm   = wv >> 1;     // 0..3 (M sub-tile)
  const int wn   = wv & 1;      // 0..1 (N sub-tile)
  const int tileM = blockIdx.y * 128;
  const int tileN = blockIdx.x * 128;
  const int lr  = lane & 15;
  const int sub = lane >> 4;

  v8f acc[2][4] = {};

  // Hoisted per-thread staging bases (loop-invariant in k0).
  size_t aBase[2];
  int    aLds[2];
  {
    #pragma unroll
    for (int it = 0; it < 2; it++) {
      int c = tid + it * 256;           // c < 512
      int i = c >> 2, ch = c & 3;
      int m = tileM + i;
      aBase[it] = (size_t)(m / MB) * (size_t)sB +
                  (size_t)(m % MB) * (size_t)sM + (size_t)(ch * 8);
      aLds[it]  = i * 32 + ch * 8;
    }
  }

  for (int k0 = 0; k0 < K; k0 += 32) {
    // ---- stage A tile [128][32] ----
    if (sK == 1) {
#if USE_ASYNC_LDS
      #pragma unroll
      for (int it = 0; it < 2; it++) {
        __builtin_amdgcn_global_load_async_to_lds_b128(
            (gas_v4i_t)(A + aBase[it] + (size_t)k0),
            (las_v4i_t)(&lA[0][0] + aLds[it]), 0, 0);
      }
#else
      #pragma unroll
      for (int it = 0; it < 2; it++) {
        *(uint4*)(&lA[0][0] + aLds[it]) = *(const uint4*)(A + aBase[it] + (size_t)k0);
      }
#endif
    } else {
      for (int e = tid; e < 4096; e += 256) {
        int i = e >> 5, kk = e & 31;
        int m = tileM + i;
        size_t addr = (size_t)(m / MB) * (size_t)sB +
                      (size_t)(m % MB) * (size_t)sM + (size_t)(k0 + kk) * (size_t)sK;
        lA[i][kk] = A[addr];
      }
    }
    // ---- stage B tile transposed into lBt[n][k] ----
    for (int c = tid; c < 512; c += 256) {
      int kk = c >> 4, jc = c & 15;
      int j0 = jc * 8;
      const __bf16* src = Bm + (size_t)(k0 + kk) * (size_t)N + (size_t)(tileN + j0);
      uint4 tmp = *(const uint4*)src;
      const __bf16* tp = (const __bf16*)&tmp;
      #pragma unroll
      for (int e = 0; e < 8; e++) lBt[j0 + e][kk] = tp[e];
    }
    if (k0 + 32 < K)
      __builtin_prefetch(Bm + (size_t)(k0 + 32) * (size_t)N + tileN, 0, 1);
#if USE_ASYNC_LDS
    if (sK == 1) __builtin_amdgcn_s_wait_asynccnt(0);
#endif
    __syncthreads();

    // ---- per-lane fragments (16-bit A layout: lanes 0-15 K0..7,16..23; 16-31 K8..15,24..31)
    FragBF a0, a1;
    {
      const __bf16* ap0 = &lA[wm * 32 + lr][sub * 8];
      a0.q[0] = *(const uint4*)ap0;
      a0.q[1] = *(const uint4*)(ap0 + 16);
      const __bf16* ap1 = &lA[wm * 32 + 16 + lr][sub * 8];
      a1.q[0] = *(const uint4*)ap1;
      a1.q[1] = *(const uint4*)(ap1 + 16);
    }
    #pragma unroll
    for (int ni = 0; ni < 4; ni++) {
      FragBF bb;
      const __bf16* bp = &lBt[wn * 64 + ni * 16 + lr][sub * 8];
      bb.q[0] = *(const uint4*)bp;
      bb.q[1] = *(const uint4*)(bp + 16);
      acc[0][ni] = WMMA_BF16(a0.v, bb.v, acc[0][ni]);
      acc[1][ni] = WMMA_BF16(a1.v, bb.v, acc[1][ni]);
    }
    __syncthreads();
  }

  // ---- epilogue: C layout: vgpr r, lane l -> M = r + (l>=16?8:0), N = l&15
  #pragma unroll
  for (int mi = 0; mi < 2; mi++) {
    #pragma unroll
    for (int ni = 0; ni < 4; ni++) {
      #pragma unroll
      for (int r = 0; r < 8; r++) {
        int row = tileM + wm * 32 + mi * 16 + r + (sub ? 8 : 0);
        int col = tileN + wn * 64 + ni * 16 + lr;
        float v = acc[mi][ni][r];
        if (bias) v += bias[col];
        if (relu) v = fmaxf(v, 0.0f);
        if (outbf16) ((__bf16*)C)[(size_t)row * N + col] = (__bf16)v;
        else         ((float*) C)[(size_t)row * N + col] = v;
      }
    }
  }
}

// ------------------------------ beam search --------------------------------
// One workgroup (256 thr = 8 waves) per (b,s) site. 7 turns.
// Layer1: [16,1024]bf16 @ [1024,512]bf16 via WMMA, wave w owns cols w*64..+63.

__global__ __launch_bounds__(256) void beam_kernel(
    const float* __restrict__ states0, const __bf16* __restrict__ goal,
    const __bf16* __restrict__ w1t, const float* __restrict__ b1,
    const float* __restrict__ w2, const float* __restrict__ b2,
    const float* __restrict__ action, const float* __restrict__ lng,
    const float* __restrict__ lnb, float* __restrict__ out)
{
  __shared__ alignas(16) unsigned char shRaw[32768];   // A[16][1024] bf16 | h[16][512] f32
  __bf16 (*shA)[1024] = (__bf16(*)[1024])shRaw;
  float  (*shH)[512]  = (float(*)[512])shRaw;
  __shared__ float sSt[10][512];
  __shared__ float sProb[10];
  __shared__ int   sHist[10][7];
  __shared__ float sLogit[16][8];
  __shared__ float sComb[80];
  __shared__ int   sNB[10], sNA[10];
  __shared__ float sNP[10];
  __shared__ int   sBest;

  const int site = blockIdx.x;
  const int b = site >> 11;
  const int s = site & 2047;
  const int tid  = threadIdx.x;
  const int lane = tid & 31;
  const int wv   = tid >> 5;
  const int lr   = lane & 15;
  const int sub  = lane >> 4;

  for (int d = tid; d < 512; d += 256) sSt[0][d] = states0[(size_t)site * 512 + d];
  if (tid == 0) sProb[0] = 1.0f;
  if (tid < 70) sHist[tid / 7][tid % 7] = 0;
  __syncthreads();

  int beam = 1;
  for (int t = 0; t < 7; t++) {
    int beam_next = beam * 8; if (beam_next > 10) beam_next = 10;

    // build A = [states | goal], pad rows >= beam with zero
    for (int idx = tid; idx < 16 * 1024; idx += 256) {
      int m = idx >> 10, j = idx & 1023;
      __bf16 v;
      if (m < beam) v = (j < 512) ? (__bf16)sSt[m][j] : goal[(size_t)site * 512 + (j - 512)];
      else          v = (__bf16)0.0f;
      shA[m][j] = v;
    }
    __syncthreads();

    v8f acc[4] = {};
    for (int k0 = 0; k0 < 1024; k0 += 32) {
      FragBF a;
      const __bf16* ap = &shA[lr][k0 + sub * 8];
      a.q[0] = *(const uint4*)ap;
      a.q[1] = *(const uint4*)(ap + 16);
      #pragma unroll
      for (int ni = 0; ni < 4; ni++) {
        int n = wv * 64 + ni * 16 + lr;
        const __bf16* bp = w1t + (size_t)n * 1024 + k0 + sub * 8;
        FragBF bb;
        bb.q[0] = *(const uint4*)bp;
        bb.q[1] = *(const uint4*)(bp + 16);
        acc[ni] = WMMA_BF16(a.v, bb.v, acc[ni]);
      }
    }
    __syncthreads();   // done reading shA; reuse as shH

    #pragma unroll
    for (int ni = 0; ni < 4; ni++) {
      #pragma unroll
      for (int r = 0; r < 8; r++) {
        int row = r + (sub ? 8 : 0);
        int col = wv * 64 + ni * 16 + lr;
        shH[row][col] = fmaxf(acc[ni][r] + b1[col], 0.0f);
      }
    }
    __syncthreads();

    // layer2: logits[m][a] = b2[a] + relu_h[m] . w2[:,a]
    if (tid < 128) {
      int m = tid >> 3, a = tid & 7;
      if (m < beam) {
        float acc2 = b2[a];
        for (int n = 0; n < 512; n++) acc2 += shH[m][n] * w2[n * 8 + a];
        sLogit[m][a] = acc2;
      }
    }
    __syncthreads();

    // softmax * states_prob
    if (tid < beam) {
      int m = tid;
      float mx = -3.4e38f;
      #pragma unroll
      for (int a = 0; a < 8; a++) mx = fmaxf(mx, sLogit[m][a]);
      float e[8], sum = 0.0f;
      #pragma unroll
      for (int a = 0; a < 8; a++) { e[a] = __expf(sLogit[m][a] - mx); sum += e[a]; }
      float p = sProb[m] / sum;
      #pragma unroll
      for (int a = 0; a < 8; a++) sComb[m * 8 + a] = p * e[a];
    }
    __syncthreads();

    // serial top-k (<=80 choose <=10) + history propagation
    if (tid == 0) {
      int n = beam * 8;
      float tmp[80];
      for (int i = 0; i < n; i++) tmp[i] = sComb[i];
      for (int j = 0; j < beam_next; j++) {
        int bi = 0; float bv = tmp[0];
        for (int i = 1; i < n; i++) if (tmp[i] > bv) { bv = tmp[i]; bi = i; }
        sNP[j] = bv; sNB[j] = bi >> 3; sNA[j] = bi & 7;
        tmp[bi] = -3.4e38f;
      }
      int hnew[10][7];
      for (int j = 0; j < beam_next; j++) {
        for (int tt = 0; tt < 7; tt++) hnew[j][tt] = sHist[sNB[j]][tt];
        hnew[j][t] = sNA[j];
      }
      for (int j = 0; j < beam_next; j++) {
        for (int tt = 0; tt < 7; tt++) sHist[j][tt] = hnew[j][tt];
        sProb[j] = sNP[j];
      }
    }
    __syncthreads();

    // gather + add action into shH
    for (int idx = tid; idx < beam_next * 512; idx += 256) {
      int j = idx >> 9, d = idx & 511;
      shH[j][d] = sSt[sNB[j]][d] + action[(size_t)sNA[j] * 512 + d];
    }
    __syncthreads();

    // layernorm rows -> sSt (one wave per row)
    for (int rep = 0; rep < 2; rep++) {
      int j = wv + rep * 8;
      if (j < beam_next) {
        float sum = 0.0f, sq = 0.0f;
        for (int d = lane; d < 512; d += 32) { float v = shH[j][d]; sum += v; sq += v * v; }
        #pragma unroll
        for (int off = 16; off > 0; off >>= 1) {
          sum += __shfl_xor(sum, off, 32);
          sq  += __shfl_xor(sq,  off, 32);
        }
        float mu  = sum * (1.0f / 512.0f);
        float var = sq * (1.0f / 512.0f) - mu * mu;
        float inv = rsqrtf(var + 1e-5f);
        for (int d = lane; d < 512; d += 32)
          sSt[j][d] = (shH[j][d] - mu) * inv * lng[d] + lnb[d];
      }
    }
    __syncthreads();
    beam = beam_next;
  }

  // final: best beam -> act0 = history[best][0]; base = initial_states (beam clip->0)
  if (tid == 0) {
    int bi = 0; float bv = sProb[0];
    for (int i = 1; i < 10; i++) if (sProb[i] > bv) { bv = sProb[i]; bi = i; }
    sBest = bi;
  }
  __syncthreads();
  int act0 = sHist[sBest][0];

  for (int d = tid; d < 512; d += 256)
    shH[0][d] = states0[(size_t)site * 512 + d] + action[(size_t)act0 * 512 + d];
  __syncthreads();

  if (wv == 0) {
    float sum = 0.0f, sq = 0.0f;
    for (int d = lane; d < 512; d += 32) { float v = shH[0][d]; sum += v; sq += v * v; }
    #pragma unroll
    for (int off = 16; off > 0; off >>= 1) {
      sum += __shfl_xor(sum, off, 32);
      sq  += __shfl_xor(sq,  off, 32);
    }
    float mu  = sum * (1.0f / 512.0f);
    float var = sq * (1.0f / 512.0f) - mu * mu;
    float inv = rsqrtf(var + 1e-5f);
    for (int d = lane; d < 512; d += 32)
      out[((size_t)b * 512 + d) * 2048 + s] = (shH[0][d] - mu) * inv * lng[d] + lnb[d];
  }
}

// ------------------------------ host launcher ------------------------------

extern "C" void kernel_launch(void* const* d_in, const int* in_sizes, int n_in,
                              void* d_out, int out_size, void* d_ws, size_t ws_size,
                              hipStream_t stream) {
  (void)in_sizes; (void)n_in; (void)out_size; (void)ws_size;
  const float* x      = (const float*)d_in[0];
  const float* weight = (const float*)d_in[1];
  const float* c2s_w1 = (const float*)d_in[2];
  const float* c2s_b1 = (const float*)d_in[3];
  const float* c2s_w2 = (const float*)d_in[4];
  const float* c2s_b2 = (const float*)d_in[5];
  const float* eg_w1  = (const float*)d_in[6];
  const float* eg_b1  = (const float*)d_in[7];
  const float* eg_w2  = (const float*)d_in[8];
  const float* eg_b2  = (const float*)d_in[9];
  const float* gna_w1 = (const float*)d_in[10];
  const float* gna_b1 = (const float*)d_in[11];
  const float* gna_w2 = (const float*)d_in[12];
  const float* gna_b2 = (const float*)d_in[13];
  const float* action = (const float*)d_in[14];
  const float* ln_g   = (const float*)d_in[15];
  const float* ln_b   = (const float*)d_in[16];
  float* out = (float*)d_out;

  char* ws = (char*)d_ws;
  size_t off = 0;
  auto alloc = [&](size_t bytes) -> char* {
    char* p = ws + off;
    off += (bytes + 255) & ~(size_t)255;
    return p;
  };

  const size_t MN = 2048u * 2048u;     // 4.19M
  const size_t TD = 8192u * 512u;      // 4.19M
  __bf16* xb   = (__bf16*)alloc(MN * 2);
  __bf16* w0b  = (__bf16*)alloc(MN * 2);
  __bf16* cb   = (__bf16*)alloc(MN * 2);
  __bf16* hb   = (__bf16*)alloc(TD * 2);
  float*  st0  = (float*) alloc(TD * 4);
  __bf16* glb  = (__bf16*)alloc(TD * 2);
  __bf16* cw1b = (__bf16*)alloc(512u * 512u * 2);
  __bf16* cw2b = (__bf16*)alloc(512u * 512u * 2);
  __bf16* ew1b = (__bf16*)alloc(512u * 512u * 2);
  __bf16* ew2b = (__bf16*)alloc(512u * 512u * 2);
  __bf16* w1tb = (__bf16*)alloc(1024u * 512u * 2);

  // conversions
  k_cvt_bf16<<<(int)(MN / 256), 256, 0, stream>>>(x, xb, (int)MN);
  k_w0_tril <<<(int)(MN / 256), 256, 0, stream>>>(weight, w0b);
  k_cvt_bf16<<<1024, 256, 0, stream>>>(c2s_w1, cw1b, 512 * 512);
  k_cvt_bf16<<<1024, 256, 0, stream>>>(c2s_w2, cw2b, 512 * 512);
  k_cvt_bf16<<<1024, 256, 0, stream>>>(eg_w1,  ew1b, 512 * 512);
  k_cvt_bf16<<<1024, 256, 0, stream>>>(eg_w2,  ew2b, 512 * 512);
  k_w1t     <<<2048, 256, 0, stream>>>(gna_w1, w1tb);

  // G1: c = relu(x @ tril(w0))  [2048x2048x2048] -> bf16
  gemm_bf16_kernel<<<dim3(16, 16), 256, 0, stream>>>(
      xb, w0b, nullptr, cb, 2048, 2048, 2048, 2048, 0L, 2048L, 1L, 1, 1);

  // G2: h1 = relu(ct @ c2s_w1 + b1)   (A = c transposed per batch)
  gemm_bf16_kernel<<<dim3(4, 64), 256, 0, stream>>>(
      cb, cw1b, c2s_b1, hb, 8192, 512, 512, 2048, 1048576L, 1L, 2048L, 1, 1);
  // G3: states = h1 @ c2s_w2 + b2  -> f32
  gemm_bf16_kernel<<<dim3(4, 64), 256, 0, stream>>>(
      hb, cw2b, c2s_b2, st0, 8192, 512, 512, 8192, 0L, 512L, 1L, 0, 0);
  // G4: g1 = relu(ct @ eg_w1 + b1)
  gemm_bf16_kernel<<<dim3(4, 64), 256, 0, stream>>>(
      cb, ew1b, eg_b1, hb, 8192, 512, 512, 2048, 1048576L, 1L, 2048L, 1, 1);
  // G5: goal = g1 @ eg_w2 + b2 -> bf16
  gemm_bf16_kernel<<<dim3(4, 64), 256, 0, stream>>>(
      hb, ew2b, eg_b2, glb, 8192, 512, 512, 8192, 0L, 512L, 1L, 0, 1);

  // beam search + final output
  beam_kernel<<<8192, 256, 0, stream>>>(
      st0, glb, w1tb, gna_b1, gna_w2, gna_b2, action, ln_g, ln_b, out);
}
